// MultiHeadAttention_50371376448063
// MI455X (gfx1250) — compile-verified, matrix operand order checked by reading, not run on hardware
//
#include <hip/hip_runtime.h>
#include <stdint.h>

#define B_  4
#define S_  2048
#define E_  768
#define H_  12
#define D_  64
#define E3_ (3 * E_)

typedef __attribute__((ext_vector_type(16))) __bf16 v16bf;
typedef __attribute__((ext_vector_type(8)))  float  v8f;

union BF2 { uint32_t u; __bf16 h[2]; };
union BF8 { uint4 u;    __bf16 h[8]; };

// K index of the low element of VGPR pair i for a 16-bit A/B WMMA fragment
// (wave32 layout: VGPRs 0-3 cover K 0..15 split by lane-half, VGPRs 4-7 cover K 16..31)
__device__ __forceinline__ int kpair_base(int i, int hl) {
  return ((i & 4) << 2) | ((i & 3) << 1) | (hl << 3);
}

__device__ __forceinline__ v16bf load_frag_lds(const __bf16* rowp, int hl, int koff) {
  v16bf f;
#pragma unroll
  for (int i = 0; i < 8; ++i) {
    BF2 p;
    p.u = *(const uint32_t*)(rowp + kpair_base(i, hl) + koff);
    f[2 * i]     = p.h[0];
    f[2 * i + 1] = p.h[1];
  }
  return f;
}

__device__ __forceinline__ v8f zero8() {
  v8f z = {0.f, 0.f, 0.f, 0.f, 0.f, 0.f, 0.f, 0.f};
  return z;
}

// ---- DPP16 butterfly reductions over a 16-lane half-row (no LDS round trips)
// quad_perm(1,0,3,2)=0xB1 (xor1), quad_perm(2,3,0,1)=0x4E (xor2),
// row_half_mirror=0x141 (lane^7 -> xor4 once quads uniform),
// row_mirror=0x140 (lane^15 -> xor8 once octets uniform).
template <int CTRL>
__device__ __forceinline__ float dpp_mov_f32(float x) {
  int v = __builtin_amdgcn_update_dpp(0, __builtin_bit_cast(int, x),
                                      CTRL, 0xf, 0xf, true);
  return __builtin_bit_cast(float, v);
}
__device__ __forceinline__ float red_max16(float x) {
  x = fmaxf(x, dpp_mov_f32<0xB1>(x));
  x = fmaxf(x, dpp_mov_f32<0x4E>(x));
  x = fmaxf(x, dpp_mov_f32<0x141>(x));
  x = fmaxf(x, dpp_mov_f32<0x140>(x));
  return x;
}
__device__ __forceinline__ float red_sum16(float x) {
  x += dpp_mov_f32<0xB1>(x);
  x += dpp_mov_f32<0x4E>(x);
  x += dpp_mov_f32<0x141>(x);
  x += dpp_mov_f32<0x140>(x);
  return x;
}

// ---------------------------------------------------------------------------
// fp32 -> bf16 downconvert (vectorized x4)
// ---------------------------------------------------------------------------
__global__ __launch_bounds__(256)
void cvt_f32_bf16(const float* __restrict__ src, __bf16* __restrict__ dst, int n4) {
  int i = blockIdx.x * 256 + threadIdx.x;
  if (i >= n4) return;
  float4 v = ((const float4*)src)[i];
  union { uint2 u; __bf16 h[4]; } t;
  t.h[0] = (__bf16)v.x; t.h[1] = (__bf16)v.y;
  t.h[2] = (__bf16)v.z; t.h[3] = (__bf16)v.w;
  ((uint2*)dst)[i] = t.u;
}

// ---------------------------------------------------------------------------
// C[M,N] = A[M,K](bf16) * B[K,N](bf16) + bias[N], bf16 or f32 output.
// 256 threads = 8 waves; block tile 128x128; wave tile 64x32 (4x2 WMMA frags).
// ---------------------------------------------------------------------------
template <bool OUT_BF16>
__global__ __launch_bounds__(256)
void gemm_bf16_wmma(const __bf16* __restrict__ A, const __bf16* __restrict__ Bm,
                    const float* __restrict__ bias, void* __restrict__ Cout,
                    int M, int N, int K) {
  __shared__ __bf16 Alds[128 * 40];   // [row][k], stride 40 (80B rows, conflict-free)
  __shared__ __bf16 Blds[128 * 40];   // transposed: [n][k], stride 40

  const int tid = threadIdx.x;
  const int lane = tid & 31;
  const int wid = tid >> 5;
  const int m16 = lane & 15;
  const int hl  = lane >> 4;
  const int waveM = (wid & 1) * 64;
  const int waveN = (wid >> 1) * 32;
  const int blockM = blockIdx.y * 128;
  const int blockN = blockIdx.x * 128;

  v8f c[4][2];
#pragma unroll
  for (int mf = 0; mf < 4; ++mf)
#pragma unroll
    for (int nf = 0; nf < 2; ++nf) c[mf][nf] = zero8();

  for (int k0 = 0; k0 < K; k0 += 32) {
    __syncthreads();
    // A tile 128x32 (uint4 loads, 2 per thread)
    for (int idx = tid; idx < 512; idx += 256) {
      int row = idx >> 2, ch = idx & 3;
      const __bf16* g = A + (size_t)(blockM + row) * K + k0 + ch * 8;
      *(uint4*)&Alds[row * 40 + ch * 8] = *(const uint4*)g;
      if (k0 + 32 < K) __builtin_prefetch(g + 32, 0, 1);
    }
    // B tile 32x128, stored transposed so K-pairs are contiguous for fragments
    for (int idx = tid; idx < 512; idx += 256) {
      int k = idx >> 4, ch = idx & 15;
      const __bf16* g = Bm + (size_t)(k0 + k) * N + blockN + ch * 8;
      BF8 t; t.u = *(const uint4*)g;
      if (k0 + 32 < K) __builtin_prefetch(g + (size_t)32 * N, 0, 1);
#pragma unroll
      for (int j = 0; j < 8; ++j) Blds[(ch * 8 + j) * 40 + k] = t.h[j];
    }
    __syncthreads();

    v16bf af[4], bfr[2];
#pragma unroll
    for (int mf = 0; mf < 4; ++mf)
      af[mf] = load_frag_lds(&Alds[(waveM + mf * 16 + m16) * 40], hl, 0);
#pragma unroll
    for (int nf = 0; nf < 2; ++nf)
      bfr[nf] = load_frag_lds(&Blds[(waveN + nf * 16 + m16) * 40], hl, 0);
#pragma unroll
    for (int mf = 0; mf < 4; ++mf)
#pragma unroll
      for (int nf = 0; nf < 2; ++nf)
        c[mf][nf] = __builtin_amdgcn_wmma_f32_16x16x32_bf16(
            false, af[mf], false, bfr[nf], (short)0, c[mf][nf], false, false);
  }

  // epilogue: C-fragment layout = lane col n, rows r + 8*half
#pragma unroll
  for (int nf = 0; nf < 2; ++nf) {
    int col = blockN + waveN + nf * 16 + m16;
    float bv = bias[col];
#pragma unroll
    for (int mf = 0; mf < 4; ++mf) {
#pragma unroll
      for (int r = 0; r < 8; ++r) {
        int row = blockM + waveM + mf * 16 + r + 8 * hl;
        float v = c[mf][nf][r] + bv;
        if (OUT_BF16) ((__bf16*)Cout)[(size_t)row * N + col] = (__bf16)v;
        else          ((float*)Cout)[(size_t)row * N + col]  = v;
      }
    }
  }
}

// ---------------------------------------------------------------------------
// Flash attention over bf16 QKV ([b*S+s, which*E + h*64 + d] layout).
// 128 threads = 4 waves; block = (q-block of 64, head, batch); wave = 16 q rows.
// ---------------------------------------------------------------------------
__global__ __launch_bounds__(128)
void mha_flash_wmma(const __bf16* __restrict__ qkv, __bf16* __restrict__ ctx) {
  __shared__ __bf16 Klds[64 * 72];      // [kv][d]
  __shared__ __bf16 Vtlds[64 * 68];     // transposed: [d][kv]
  __shared__ __bf16 Plds[4][16 * 68];   // per-wave P staging (C-layout -> A-layout)

  const int tid  = threadIdx.x;
  const int lane = tid & 31;
  const int wid  = tid >> 5;
  const int m16  = lane & 15;
  const int hl   = lane >> 4;
  const int b = blockIdx.z, h = blockIdx.y;
  const int qrow0 = blockIdx.x * 64 + wid * 16;

  const __bf16* base = qkv + (size_t)b * S_ * E3_ + h * D_;

  // Q fragments, softmax scale D^-0.5 = 0.125 folded in (exact in bf16)
  v16bf qf[2];
#pragma unroll
  for (int kf = 0; kf < 2; ++kf) {
#pragma unroll
    for (int i = 0; i < 8; ++i) {
      int d = kpair_base(i, hl) + kf * 32;
      BF2 p;
      p.u = *(const uint32_t*)(base + (size_t)(qrow0 + m16) * E3_ + d);
      qf[kf][2 * i]     = (__bf16)((float)p.h[0] * 0.125f);
      qf[kf][2 * i + 1] = (__bf16)((float)p.h[1] * 0.125f);
    }
  }

  v8f o[4];
#pragma unroll
  for (int ds = 0; ds < 4; ++ds) o[ds] = zero8();
  float mrow[8], lrow[8];
#pragma unroll
  for (int r = 0; r < 8; ++r) { mrow[r] = -__builtin_inff(); lrow[r] = 0.f; }

  for (int kv0 = 0; kv0 < S_; kv0 += 64) {
    __syncthreads();
    // K tile 64x64 (coalesced uint4)
    for (int idx = tid; idx < 512; idx += 128) {
      int kr = idx >> 3, ch = idx & 7;
      const __bf16* g = base + (size_t)(kv0 + kr) * E3_ + E_ + ch * 8;
      *(uint4*)&Klds[kr * 72 + ch * 8] = *(const uint4*)g;
    }
    // V tile 64x64, transposed so kv-pairs are contiguous for B fragments
    for (int idx = tid; idx < 512; idx += 128) {
      int kr = idx >> 3, ch = idx & 7;
      const __bf16* g = base + (size_t)(kv0 + kr) * E3_ + 2 * E_ + ch * 8;
      BF8 t; t.u = *(const uint4*)g;
#pragma unroll
      for (int j = 0; j < 8; ++j) Vtlds[(ch * 8 + j) * 68 + kr] = t.h[j];
    }
    __syncthreads();

    // S = (Q*scale) K^T : 16x64 per wave, 8 WMMAs
    v8f s4[4];
#pragma unroll
    for (int sub = 0; sub < 4; ++sub) {
      s4[sub] = zero8();
#pragma unroll
      for (int kf = 0; kf < 2; ++kf) {
        v16bf kfrag = load_frag_lds(&Klds[(sub * 16 + m16) * 72], hl, kf * 32);
        s4[sub] = __builtin_amdgcn_wmma_f32_16x16x32_bf16(
            false, qf[kf], false, kfrag, (short)0, s4[sub], false, false);
      }
    }

    // online softmax; row r+8*hl values live in one 16-lane half -> DPP16 butterfly
    float mx[8];
#pragma unroll
    for (int r = 0; r < 8; ++r) {
      mx[r] = fmaxf(fmaxf(s4[0][r], s4[1][r]), fmaxf(s4[2][r], s4[3][r]));
      mx[r] = red_max16(mx[r]);
    }

    float alpha[8], rs[8];
#pragma unroll
    for (int r = 0; r < 8; ++r) {
      float mn = fmaxf(mrow[r], mx[r]);
      alpha[r] = __expf(mrow[r] - mn);
      mrow[r] = mn;
      rs[r] = 0.f;
    }
#pragma unroll
    for (int sub = 0; sub < 4; ++sub)
#pragma unroll
      for (int r = 0; r < 8; ++r) {
        float p = __expf(s4[sub][r] - mrow[r]);
        s4[sub][r] = p;
        rs[r] += p;
      }
#pragma unroll
    for (int r = 0; r < 8; ++r) {
      rs[r] = red_sum16(rs[r]);
      lrow[r] = lrow[r] * alpha[r] + rs[r];
    }
#pragma unroll
    for (int ds = 0; ds < 4; ++ds)
#pragma unroll
      for (int r = 0; r < 8; ++r) o[ds][r] *= alpha[r];

    // P -> wave-private LDS (same-wave DS ordering makes this safe w/o barrier)
    __bf16* P = &Plds[wid][0];
#pragma unroll
    for (int sub = 0; sub < 4; ++sub)
#pragma unroll
      for (int r = 0; r < 8; ++r)
        P[(r + 8 * hl) * 68 + sub * 16 + m16] = (__bf16)s4[sub][r];

    // O += P V : 8 WMMAs
#pragma unroll
    for (int kf = 0; kf < 2; ++kf) {
      v16bf pf = load_frag_lds(&P[m16 * 68], hl, kf * 32);
#pragma unroll
      for (int ds = 0; ds < 4; ++ds) {
        v16bf vf = load_frag_lds(&Vtlds[(ds * 16 + m16) * 68], hl, kf * 32);
        o[ds] = __builtin_amdgcn_wmma_f32_16x16x32_bf16(
            false, pf, false, vf, (short)0, o[ds], false, false);
      }
    }
  }

  // normalize and store context as bf16 in [b*S+s, h*64+d] layout
#pragma unroll
  for (int ds = 0; ds < 4; ++ds)
#pragma unroll
    for (int r = 0; r < 8; ++r) {
      float v = o[ds][r] / lrow[r];
      int srow = qrow0 + r + 8 * hl;
      ctx[(size_t)(b * S_ + srow) * E_ + h * D_ + ds * 16 + m16] = (__bf16)v;
    }
}

// ---------------------------------------------------------------------------
extern "C" void kernel_launch(void* const* d_in, const int* in_sizes, int n_in,
                              void* d_out, int out_size, void* d_ws, size_t ws_size,
                              hipStream_t stream) {
  (void)in_sizes; (void)n_in; (void)out_size; (void)ws_size;
  const float* x      = (const float*)d_in[0];
  const float* W_qkv  = (const float*)d_in[1];
  const float* b_qkv  = (const float*)d_in[2];
  const float* W_proj = (const float*)d_in[3];
  const float* b_proj = (const float*)d_in[4];

  const size_t M = (size_t)B_ * S_;   // 8192
  char* ws = (char*)d_ws;
  __bf16* x_bf   = (__bf16*)ws;  ws += M * E_ * 2;                 // 12.6 MB
  __bf16* wq_bf  = (__bf16*)ws;  ws += (size_t)E_ * E3_ * 2;       //  3.5 MB
  __bf16* wp_bf  = (__bf16*)ws;  ws += (size_t)E_ * E_ * 2;        //  1.2 MB
  __bf16* qkv_bf = (__bf16*)ws;  ws += M * E3_ * 2;                // 37.7 MB
  __bf16* ctx_bf = (__bf16*)ws;                                    // 12.6 MB

  {
    int n4 = (int)(M * E_ / 4);
    hipLaunchKernelGGL(cvt_f32_bf16, dim3((n4 + 255) / 256), dim3(256), 0, stream, x, x_bf, n4);
    n4 = (int)((size_t)E_ * E3_ / 4);
    hipLaunchKernelGGL(cvt_f32_bf16, dim3((n4 + 255) / 256), dim3(256), 0, stream, W_qkv, wq_bf, n4);
    n4 = (int)((size_t)E_ * E_ / 4);
    hipLaunchKernelGGL(cvt_f32_bf16, dim3((n4 + 255) / 256), dim3(256), 0, stream, W_proj, wp_bf, n4);
  }

  // QKV = x @ W_qkv + b_qkv   (8192 x 2304 x 768), bf16 out
  hipLaunchKernelGGL((gemm_bf16_wmma<true>), dim3(E3_ / 128, (int)(M / 128)), dim3(256), 0,
                     stream, x_bf, wq_bf, b_qkv, (void*)qkv_bf, (int)M, E3_, E_);

  // Flash attention per (q-block, head, batch)
  hipLaunchKernelGGL(mha_flash_wmma, dim3(S_ / 64, H_, B_), dim3(128), 0, stream,
                     qkv_bf, ctx_bf);

  // out = ctx @ W_proj + b_proj  (8192 x 768 x 768), fp32 out
  hipLaunchKernelGGL((gemm_bf16_wmma<false>), dim3(E_ / 128, (int)(M / 128)), dim3(256), 0,
                     stream, ctx_bf, wp_bf, b_proj, d_out, (int)M, E_, E_);
}
